// DecoderRNN_50938312131021
// MI455X (gfx1250) — matrix operand order repaired and checked by reading, not compiled, not run on hardware
//
#include <hip/hip_runtime.h>
#include <hip/hip_bf16.h>

#define Hn      64
#define OUT_N   4761
#define T_STEPS 4096
#define NSEQ    2
#define ROWS    (T_STEPS * NSEQ)   // 8192 total projected vectors
#define CTILES  300                 // ceil(4800/16) column tiles of 16
#define OUTP    (CTILES * 16)       // 4800 padded output columns

typedef __attribute__((ext_vector_type(16))) _Float16 v16h;
typedef __attribute__((ext_vector_type(8)))  _Float16 v8h;
typedef __attribute__((ext_vector_type(8)))  float    v8f;

// ---------------------------------------------------------------------------
// Kernel 1: convert W_lin (OUT_N x 64, f32 row-major) into f16, pre-swizzled
// into a per-lane WMMA B-operand layout for v_wmma_f32_16x16x32_f16 (contiguous k, same as A).
// Layout: Bsw[((ct*2 + kb)*32 + lane)*16 + e], where for lane L (N = ct*16 +
// (L&15), half = L>>4) element e maps to K = kb*32 + half*16 + e.
// Padded columns (n >= OUT_N) are zero.
// ---------------------------------------------------------------------------
__global__ void pack_b_kernel(const float* __restrict__ W_lin,
                              _Float16* __restrict__ Bsw) {
    int tid = blockIdx.x * blockDim.x + threadIdx.x;
    if (tid >= CTILES * 2 * 32 * 16) return;
    int e  = tid & 15;
    int L  = (tid >> 4) & 31;
    int kb = (tid >> 9) & 1;
    int ct = tid >> 10;
    int n  = ct * 16 + (L & 15);
    int hs = L >> 4;
    int k  = kb * 32 + hs * 16 + e;
    float v = (n < OUT_N) ? W_lin[n * Hn + k] : 0.0f;
    Bsw[tid] = (_Float16)v;
}

// ---------------------------------------------------------------------------
// Kernel 2: the serial 2-layer tanh RNN recurrence. One workgroup, 64 threads
// (2 wave32s co-executing). Thread i owns output row i of each weight matrix,
// kept entirely in VGPRs (3*64 floats). The hidden vectors live in LDS and
// are read with broadcast b128 loads. Top-layer outputs stored as f16 to
// Hs[(s*T_STEPS + t)*64 + i] so the final GEMM output is row-linear.
// ---------------------------------------------------------------------------
__global__ __launch_bounds__(64) void rnn_recurrence_kernel(
    const float* __restrict__ hidden,
    const float* __restrict__ W_hh0,
    const float* __restrict__ b_ih0, const float* __restrict__ b_hh0,
    const float* __restrict__ W_ih1, const float* __restrict__ W_hh1,
    const float* __restrict__ b_ih1, const float* __restrict__ b_hh1,
    _Float16* __restrict__ Hs) {
    __shared__ __align__(16) float h0s[Hn];
    __shared__ __align__(16) float h1s[Hn];
    const int i = threadIdx.x;

    float wa[Hn], wb[Hn], wc[Hn];
#pragma unroll
    for (int j = 0; j < Hn; ++j) {
        wa[j] = W_hh0[i * Hn + j];   // layer0 recurrent row (x=0 -> W_ih0 unused)
        wb[j] = W_ih1[i * Hn + j];   // layer1 input row
        wc[j] = W_hh1[i * Hn + j];   // layer1 recurrent row
    }
    const float b0 = b_ih0[i] + b_hh0[i];
    const float b1 = b_ih1[i] + b_hh1[i];

    h0s[i] = hidden[i];
    h1s[i] = hidden[Hn + i];
    __syncthreads();

    for (int t = 0; t < T_STEPS; ++t) {
#pragma unroll
        for (int s = 0; s < NSEQ; ++s) {
            // ---- layer 0: h0 = tanh(b0 + W_hh0 . h0) ----
            float a0 = b0, a1 = 0.0f, a2 = 0.0f, a3 = 0.0f;
#pragma unroll
            for (int j = 0; j < Hn; j += 4) {
                float4 hx = *reinterpret_cast<const float4*>(&h0s[j]);
                a0 = fmaf(wa[j + 0], hx.x, a0);
                a1 = fmaf(wa[j + 1], hx.y, a1);
                a2 = fmaf(wa[j + 2], hx.z, a2);
                a3 = fmaf(wa[j + 3], hx.w, a3);
            }
            float h0new = tanhf((a0 + a1) + (a2 + a3));
            __syncthreads();            // all reads of h0s complete
            h0s[i] = h0new;
            __syncthreads();            // new h0 visible

            // ---- layer 1: h1 = tanh(b1 + W_ih1 . h0 + W_hh1 . h1) ----
            float c0 = b1, c1 = 0.0f, c2 = 0.0f, c3 = 0.0f;
#pragma unroll
            for (int j = 0; j < Hn; j += 4) {
                float4 hx = *reinterpret_cast<const float4*>(&h0s[j]);
                float4 hy = *reinterpret_cast<const float4*>(&h1s[j]);
                c0 = fmaf(wb[j + 0], hx.x, c0);
                c1 = fmaf(wb[j + 1], hx.y, c1);
                c2 = fmaf(wb[j + 2], hx.z, c2);
                c3 = fmaf(wb[j + 3], hx.w, c3);
                c0 = fmaf(wc[j + 0], hy.x, c0);
                c1 = fmaf(wc[j + 1], hy.y, c1);
                c2 = fmaf(wc[j + 2], hy.z, c2);
                c3 = fmaf(wc[j + 3], hy.w, c3);
            }
            float h1new = tanhf((c0 + c1) + (c2 + c3));
            __syncthreads();            // all reads of h1s complete
            h1s[i] = h1new;
            __syncthreads();            // new h1 visible

            Hs[((size_t)s * T_STEPS + t) * Hn + i] = (_Float16)h1new;
        }
    }
}

// ---------------------------------------------------------------------------
// Kernel 3: Out[R][n] = Hs[R][:] . W_lin[n][:] + b_lin[n], R=0..8191,
// n=0..4760, Out row-major == reference (2,4096,4761) flat. Each wave owns a
// 16x64 output tile: 4 f32 accumulators, K=64 via two chained
// v_wmma_f32_16x16x32_f16. A loads are two 16B per-lane loads; B loads are
// contiguous 32B per-lane from the pre-swizzled buffer. Non-temporal stores
// (156 MB written once — this phase is pure HBM store bandwidth, ~6.7 us).
// ---------------------------------------------------------------------------
__global__ __launch_bounds__(256) void gemm_out_kernel(
    const _Float16* __restrict__ Hs, const _Float16* __restrict__ Bsw,
    const float* __restrict__ b_lin, float* __restrict__ out) {
    const int lane = threadIdx.x & 31;
    const int wave = threadIdx.x >> 5;
    const int row0 = blockIdx.x * 16;
    const int col0 = (blockIdx.y * 8 + wave) * 64;
    if (col0 >= OUTP) return;                      // wave-uniform guard

    const int m  = lane & 15;
    const int hs = lane >> 4;

    v8f acc0 = {}, acc1 = {}, acc2 = {}, acc3 = {};
    const _Float16* arow = Hs + (size_t)(row0 + m) * Hn;

#pragma unroll
    for (int kb = 0; kb < 2; ++kb) {
        const int k0 = kb * 32;
        // A-operand lane layout: element e -> K=k0+hs*16+e
        v8h alo = *reinterpret_cast<const v8h*>(arow + k0 + hs * 16);
        v8h ahi = *reinterpret_cast<const v8h*>(arow + k0 + hs * 16 + 8);
        v16h A;
#pragma unroll
        for (int e = 0; e < 8; ++e) { A[e] = alo[e]; A[e + 8] = ahi[e]; }

        const size_t ct0 = (size_t)(col0 >> 4);
#pragma unroll
        for (int c = 0; c < 4; ++c) {
            v16h B = *reinterpret_cast<const v16h*>(
                Bsw + (((ct0 + c) * 2 + kb) * 32 + lane) * 16);
            v8f& acc = (c == 0) ? acc0 : (c == 1) ? acc1 : (c == 2) ? acc2 : acc3;
            acc = __builtin_amdgcn_wmma_f32_16x16x32_f16(
                false, A, false, B, (short)0, acc, false, false);
        }
    }

#pragma unroll
    for (int c = 0; c < 4; ++c) {
        const int n = col0 + c * 16 + m;
        if (n < OUT_N) {
            const float bias = b_lin[n];
            const v8f& acc = (c == 0) ? acc0 : (c == 1) ? acc1 : (c == 2) ? acc2 : acc3;
#pragma unroll
            for (int r = 0; r < 8; ++r) {
                const int R = row0 + r + hs * 8;   // C/D layout: VGPR r -> M=r(+8)
                __builtin_nontemporal_store(acc[r] + bias,
                                            out + (size_t)R * OUT_N + n);
            }
        }
    }
}

// ---------------------------------------------------------------------------
extern "C" void kernel_launch(void* const* d_in, const int* in_sizes, int n_in,
                              void* d_out, int out_size, void* d_ws, size_t ws_size,
                              hipStream_t stream) {
    (void)in_sizes; (void)n_in; (void)out_size; (void)ws_size;
    const float* hidden = (const float*)d_in[0];
    // d_in[1] = W_ih0: multiplied by a zero input, never needed.
    const float* W_hh0  = (const float*)d_in[2];
    const float* b_ih0  = (const float*)d_in[3];
    const float* b_hh0  = (const float*)d_in[4];
    const float* W_ih1  = (const float*)d_in[5];
    const float* W_hh1  = (const float*)d_in[6];
    const float* b_ih1  = (const float*)d_in[7];
    const float* b_hh1  = (const float*)d_in[8];
    const float* W_lin  = (const float*)d_in[9];
    const float* b_lin  = (const float*)d_in[10];
    float* out = (float*)d_out;

    _Float16* Hs  = (_Float16*)d_ws;                                   // 1 MiB
    _Float16* Bsw = (_Float16*)((char*)d_ws + (size_t)ROWS * Hn * 2);  // 600 KiB

    const int bElems = CTILES * 2 * 32 * 16;
    pack_b_kernel<<<(bElems + 255) / 256, 256, 0, stream>>>(W_lin, Bsw);

    rnn_recurrence_kernel<<<1, 64, 0, stream>>>(hidden, W_hh0, b_ih0, b_hh0,
                                                W_ih1, W_hh1, b_ih1, b_hh1, Hs);

    dim3 grid(ROWS / 16, (OUTP + 511) / 512);   // (512, 10); 8 waves/block x 64 cols
    gemm_out_kernel<<<grid, 256, 0, stream>>>(Hs, Bsw, b_lin, out);
}
